// WindowAttention_77223511982373
// MI455X (gfx1250) — matrix operand order repaired and checked by reading, not compiled, not run on hardware
//
#include <hip/hip_runtime.h>
#include <hip/hip_bf16.h>

typedef __attribute__((ext_vector_type(16))) __bf16 v16bf;
typedef __attribute__((ext_vector_type(8)))  __bf16 v8bf;
typedef __attribute__((ext_vector_type(8)))  float  v8f;
typedef __attribute__((ext_vector_type(4)))  float  v4f;

#define TOK 49
#define SCALE_Q 0.17677669529663687f  // 32^-0.5
#define ASTR 40   // padded LDS stride for A tiles (64x32): conflict-free, 16B-aligned
#define BSTR 40   // padded LDS stride for [n][k] B tiles (128x32)
#define PSTR 72   // padded LDS stride for attention P tile (64x64)

__device__ __forceinline__ __bf16 f2bf(float f) {
  unsigned u = __builtin_bit_cast(unsigned, f);
  u += 0x7fffu + ((u >> 16) & 1u);               // round-to-nearest-even
  unsigned short h = (unsigned short)(u >> 16);
  return __builtin_bit_cast(__bf16, h);
}

__device__ __forceinline__ v16bf frag_zero() {
  v16bf z;
#pragma unroll
  for (int e = 0; e < 16; ++e) z[e] = __builtin_bit_cast(__bf16, (unsigned short)0);
  return z;
}

__device__ __forceinline__ v8f acc_zero() {
  v8f z;
#pragma unroll
  for (int e = 0; e < 8; ++e) z[e] = 0.0f;
  return z;
}

__device__ __forceinline__ v8f wmma_bf16(v16bf a, v16bf b, v8f c) {
  return __builtin_amdgcn_wmma_f32_16x16x32_bf16(false, a, false, b, (short)0, c, false, false);
}

// ---------------- fp32 -> bf16, 4 elems/thread ----------------
__global__ void cvt4_f32_bf16(const float* __restrict__ src, __bf16* __restrict__ dst, int n4) {
  int i = blockIdx.x * blockDim.x + threadIdx.x;
  if (i < n4) {
    v4f v = *(const v4f*)(src + (size_t)i * 4);
#pragma unroll
    for (int j = 0; j < 4; ++j) dst[(size_t)i * 4 + j] = f2bf(v[j]);
  }
}

// ---------------- fp32 [R][C] -> bf16 transposed [C][R], compile-time dims ----------------
template <int R, int C>
__global__ void cvt_wT(const float* __restrict__ src, __bf16* __restrict__ dst) {
  int c = blockIdx.x * 256 + threadIdx.x;
  int r = blockIdx.y;
  if (c < C) dst[(size_t)c * R + r] = f2bf(src[(size_t)r * C + c]);
}

// ---------------- combined rel-bias + mask table: C[w][h][i][j] ----------------
__global__ void build_bias(const float* __restrict__ mask, const float* __restrict__ bt,
                           float* __restrict__ C) {
  int t = blockIdx.x * 256 + threadIdx.x;       // 0..2400 = i*49+j
  if (t >= TOK * TOK) return;
  int wh = blockIdx.y;                           // w*16+h
  int h = wh & 15, w = wh >> 4;
  int i = t / TOK, j = t % TOK;                  // constant-divisor (magic) division
  int d0 = i / 7 - j / 7 + 6;
  int d1 = i % 7 - j % 7 + 6;
  C[(size_t)wh * TOK * TOK + t] = bt[(d0 * 13 + d1) * 16 + h] + mask[(size_t)w * TOK * TOK + t];
}

// ---------------- zero pad columns 48..63 of vT rows ----------------
__global__ void zero_vt_pad(__bf16* __restrict__ vT, int nrows) {
  int row = blockIdx.x * blockDim.x + threadIdx.x;
  if (row < nrows) {
    v8bf z;
#pragma unroll
    for (int e = 0; e < 8; ++e) z[e] = __builtin_bit_cast(__bf16, (unsigned short)0);
    *(v8bf*)(vT + (size_t)row * 64 + 48) = z;
    *(v8bf*)(vT + (size_t)row * 64 + 56) = z;
  }
}

// ---------------- QKV GEMM: xb[100352,512] x wT[1536,512] -> Q/K [B,H,49,32], vT [B,H,32,64] ----
__global__ __launch_bounds__(128)
void qkv_gemm(const __bf16* __restrict__ xb, const __bf16* __restrict__ wT,
              const float* __restrict__ bias,
              __bf16* __restrict__ qo, __bf16* __restrict__ ko, __bf16* __restrict__ vT) {
  __shared__ __bf16 As[2 * 64 * ASTR];    // ping-pong [m][k]
  __shared__ __bf16 Bs[2 * 128 * BSTR];   // ping-pong [n][k]
  const int tid  = threadIdx.x;
  const int lane = tid & 31;
  const int wid  = tid >> 5;
  const int m0   = blockIdx.y * 64;
  const int n0   = blockIdx.x * 128;
  const int wm   = (wid >> 1) * 32;
  const int wn   = (wid & 1) * 64;
  const int ln   = lane & 15;
  const int hi   = lane >> 4;
  const int ar0 = tid >> 2;               // 0..31
  const int acg = (tid & 3) * 8;          // k group of 8
  const int bn0 = tid >> 2;               // 0..31
  const int bkg = (tid & 3) * 8;          // k group of 8

  v8f acc[2][4];
#pragma unroll
  for (int mi = 0; mi < 2; ++mi)
#pragma unroll
    for (int nj = 0; nj < 4; ++nj) acc[mi][nj] = acc_zero();

  v8bf ra[2], rb[4];
  auto load_tiles = [&](int ks) {
#pragma unroll
    for (int e = 0; e < 2; ++e)
      ra[e] = *(const v8bf*)(xb + (size_t)(m0 + ar0 + e * 32) * 512 + ks + acg);
#pragma unroll
    for (int e = 0; e < 4; ++e)
      rb[e] = *(const v8bf*)(wT + (size_t)(n0 + bn0 + e * 32) * 512 + ks + bkg);
  };
  auto store_tiles = [&](int p) {
    __bf16* Asp = As + p * (64 * ASTR);
    __bf16* Bsp = Bs + p * (128 * BSTR);
#pragma unroll
    for (int e = 0; e < 2; ++e)
#pragma unroll
      for (int j = 0; j < 8; ++j) Asp[(ar0 + e * 32) * ASTR + acg + j] = ra[e][j];
#pragma unroll
    for (int e = 0; e < 4; ++e)
#pragma unroll
      for (int j = 0; j < 8; ++j) Bsp[(bn0 + e * 32) * BSTR + bkg + j] = rb[e][j];
  };

  load_tiles(0);
  store_tiles(0);
  int p = 0;
  for (int ks = 0; ks < 512; ks += 32) {
    __syncthreads();
    const bool more = (ks + 32 < 512);
    if (more) load_tiles(ks + 32);

    const __bf16* Asp = As + p * (64 * ASTR);
    const __bf16* Bsp = Bs + p * (128 * BSTR);
    v16bf af[2], bf[4];
#pragma unroll
    for (int mi = 0; mi < 2; ++mi) {
      int m = wm + mi * 16 + ln;
#pragma unroll
      for (int e = 0; e < 16; ++e) {
        int kk = hi * 16 + e;
        af[mi][e] = Asp[m * ASTR + kk];
      }
    }
#pragma unroll
    for (int nj = 0; nj < 4; ++nj) {
      int nn = wn + nj * 16 + ln;
      int kb = hi * 16;
#pragma unroll
      for (int e = 0; e < 16; ++e) bf[nj][e] = Bsp[nn * BSTR + kb + e];
    }
#pragma unroll
    for (int mi = 0; mi < 2; ++mi)
#pragma unroll
      for (int nj = 0; nj < 4; ++nj)
        acc[mi][nj] = wmma_bf16(af[mi], bf[nj], acc[mi][nj]);

    if (more) store_tiles(p ^ 1);
    p ^= 1;
  }

  // epilogue: +bias; Q scaled; V stored transposed+padded
#pragma unroll
  for (int mi = 0; mi < 2; ++mi)
#pragma unroll
    for (int nj = 0; nj < 4; ++nj)
#pragma unroll
      for (int r = 0; r < 8; ++r) {
        int row = m0 + wm + mi * 16 + hi * 8 + r;
        int col = n0 + wn + nj * 16 + ln;
        float val = acc[mi][nj][r] + bias[col];
        int sel  = col >> 9;
        int w5   = col & 511;
        int head = w5 >> 5;
        int hd   = w5 & 31;
        int b = row / TOK, t = row % TOK;
        size_t pair = (size_t)b * 16 + head;
        if (sel == 0) {
          qo[(pair * TOK + t) * 32 + hd] = f2bf(val * SCALE_Q);
        } else if (sel == 1) {
          ko[(pair * TOK + t) * 32 + hd] = f2bf(val);
        } else {
          vT[(pair * 32 + hd) * 64 + t] = f2bf(val);
        }
      }
}

// ---------------- attention: one wave per (b,h) ----------------
__global__ __launch_bounds__(128)
void attn_kernel(const __bf16* __restrict__ q, const __bf16* __restrict__ kbuf,
                 const __bf16* __restrict__ vT, const float* __restrict__ Cb,
                 __bf16* __restrict__ ao) {
  __shared__ __bf16 Plds[4][64 * PSTR];
  const int lane = threadIdx.x & 31;
  const int wid  = threadIdx.x >> 5;
  const int pair = blockIdx.x * 4 + wid;      // = b*16 + h
  const int b = pair >> 4;
  const int h = pair & 15;
  const __bf16* qp  = q    + (size_t)pair * TOK * 32;
  const __bf16* kp  = kbuf + (size_t)pair * TOK * 32;
  const __bf16* vtp = vT   + (size_t)pair * 32 * 64;
  const float*  cb  = Cb   + ((size_t)(b & 63) * 16 + h) * TOK * TOK;
  const int ln = lane & 15;
  const int hi = lane >> 4;

  // --- S = Q * K^T (Q already scaled) ---
  v16bf qf[4];
#pragma unroll
  for (int ti = 0; ti < 4; ++ti) {
    qf[ti] = frag_zero();
    int m = ti * 16 + ln;
    if (m < TOK) {
#pragma unroll
      for (int e = 0; e < 8; ++e) qf[ti][e]     = qp[m * 32 + hi * 8 + e];
#pragma unroll
      for (int e = 0; e < 8; ++e) qf[ti][8 + e] = qp[m * 32 + 16 + hi * 8 + e];
    }
  }

  v8f s[4][4];
#pragma unroll
  for (int ti = 0; ti < 4; ++ti)
#pragma unroll
    for (int tj = 0; tj < 4; ++tj) s[ti][tj] = acc_zero();

#pragma unroll
  for (int tj = 0; tj < 4; ++tj) {
    v16bf bfr = frag_zero();
    int tok = tj * 16 + ln;               // B[k=hd][n=tok] = K[tok][hd]
    if (tok < TOK) {
      int kb = hi * 8;
#pragma unroll
      for (int e = 0; e < 16; ++e) bfr[e] = kp[tok * 32 + kb + (e >> 3) * 16 + (e & 7)];
    }
#pragma unroll
    for (int ti = 0; ti < 4; ++ti)
      s[ti][tj] = wmma_bf16(qf[ti], bfr, s[ti][tj]);
  }

  // --- +combined bias table, register softmax, P -> LDS (bf16) ---
  __bf16* P = Plds[wid];
#pragma unroll
  for (int ti = 0; ti < 4; ++ti) {
#pragma unroll
    for (int r = 0; r < 8; ++r) {
      int i = ti * 16 + hi * 8 + r;
      float vals[4];
#pragma unroll
      for (int tj = 0; tj < 4; ++tj) {
        int j = tj * 16 + ln;
        vals[tj] = (i < TOK && j < TOK) ? (s[ti][tj][r] + cb[i * TOK + j]) : -1e30f;
      }
      float mx = fmaxf(fmaxf(vals[0], vals[1]), fmaxf(vals[2], vals[3]));
#pragma unroll
      for (int xm = 1; xm < 16; xm <<= 1) mx = fmaxf(mx, __shfl_xor(mx, xm, 32));
      float sum = 0.0f;
#pragma unroll
      for (int tj = 0; tj < 4; ++tj) { vals[tj] = __expf(vals[tj] - mx); sum += vals[tj]; }
#pragma unroll
      for (int xm = 1; xm < 16; xm <<= 1) sum += __shfl_xor(sum, xm, 32);
      float inv = 1.0f / sum;
#pragma unroll
      for (int tj = 0; tj < 4; ++tj) P[i * PSTR + tj * 16 + ln] = f2bf(vals[tj] * inv);
    }
  }
  __syncthreads();

  // --- O = P * V ---
  v8f o[4][2];
#pragma unroll
  for (int mi = 0; mi < 4; ++mi)
#pragma unroll
    for (int nj = 0; nj < 2; ++nj) o[mi][nj] = acc_zero();

#pragma unroll
  for (int ss = 0; ss < 2; ++ss) {
    v16bf af[4];
#pragma unroll
    for (int mi = 0; mi < 4; ++mi) {
      int m = mi * 16 + ln;
#pragma unroll
      for (int e = 0; e < 16; ++e) {
        int kk = ss * 32 + hi * 16 + e;
        af[mi][e] = P[m * PSTR + kk];
      }
    }
#pragma unroll
    for (int nj = 0; nj < 2; ++nj) {
      int hd = nj * 16 + ln;
      // B[k=tok][n=hd] = vT[hd][tok], contiguous & zero-padded
      v16bf bfr = *(const v16bf*)(vtp + (size_t)hd * 64 + ss * 32 + hi * 16);
#pragma unroll
      for (int mi = 0; mi < 4; ++mi)
        o[mi][nj] = wmma_bf16(af[mi], bfr, o[mi][nj]);
    }
  }

  // scatter to ao[b][tok][h*32+hd] bf16
  const size_t obase = (size_t)b * TOK * 512 + h * 32;
#pragma unroll
  for (int mi = 0; mi < 4; ++mi)
#pragma unroll
    for (int nj = 0; nj < 2; ++nj)
#pragma unroll
      for (int r = 0; r < 8; ++r) {
        int i = mi * 16 + hi * 8 + r;
        if (i < TOK) ao[obase + (size_t)i * 512 + nj * 16 + ln] = f2bf(o[mi][nj][r]);
      }
}

// ---------------- proj GEMM: ao[100352,512] x wprojT[512,512] -> fp32 out ----------------
__global__ __launch_bounds__(128)
void proj_gemm(const __bf16* __restrict__ a, const __bf16* __restrict__ wT,
               const float* __restrict__ bias, float* __restrict__ out) {
  __shared__ __bf16 As[2 * 64 * ASTR];
  __shared__ __bf16 Bs[2 * 128 * BSTR];
  const int tid  = threadIdx.x;
  const int lane = tid & 31;
  const int wid  = tid >> 5;
  const int m0   = blockIdx.y * 64;
  const int n0   = blockIdx.x * 128;
  const int wm   = (wid >> 1) * 32;
  const int wn   = (wid & 1) * 64;
  const int ln   = lane & 15;
  const int hi   = lane >> 4;
  const int ar0 = tid >> 2;
  const int acg = (tid & 3) * 8;
  const int bn0 = tid >> 2;
  const int bkg = (tid & 3) * 8;

  v8f acc[2][4];
#pragma unroll
  for (int mi = 0; mi < 2; ++mi)
#pragma unroll
    for (int nj = 0; nj < 4; ++nj) acc[mi][nj] = acc_zero();

  v8bf ra[2], rb[4];
  auto load_tiles = [&](int ks) {
#pragma unroll
    for (int e = 0; e < 2; ++e)
      ra[e] = *(const v8bf*)(a + (size_t)(m0 + ar0 + e * 32) * 512 + ks + acg);
#pragma unroll
    for (int e = 0; e < 4; ++e)
      rb[e] = *(const v8bf*)(wT + (size_t)(n0 + bn0 + e * 32) * 512 + ks + bkg);
  };
  auto store_tiles = [&](int p) {
    __bf16* Asp = As + p * (64 * ASTR);
    __bf16* Bsp = Bs + p * (128 * BSTR);
#pragma unroll
    for (int e = 0; e < 2; ++e)
#pragma unroll
      for (int j = 0; j < 8; ++j) Asp[(ar0 + e * 32) * ASTR + acg + j] = ra[e][j];
#pragma unroll
    for (int e = 0; e < 4; ++e)
#pragma unroll
      for (int j = 0; j < 8; ++j) Bsp[(bn0 + e * 32) * BSTR + bkg + j] = rb[e][j];
  };

  load_tiles(0);
  store_tiles(0);
  int p = 0;
  for (int ks = 0; ks < 512; ks += 32) {
    __syncthreads();
    const bool more = (ks + 32 < 512);
    if (more) load_tiles(ks + 32);

    const __bf16* Asp = As + p * (64 * ASTR);
    const __bf16* Bsp = Bs + p * (128 * BSTR);
    v16bf af[2], bf[4];
#pragma unroll
    for (int mi = 0; mi < 2; ++mi) {
      int m = wm + mi * 16 + ln;
#pragma unroll
      for (int e = 0; e < 16; ++e) {
        int kk = hi * 16 + e;
        af[mi][e] = Asp[m * ASTR + kk];
      }
    }
#pragma unroll
    for (int nj = 0; nj < 4; ++nj) {
      int nn = wn + nj * 16 + ln;
      int kb = hi * 16;
#pragma unroll
      for (int e = 0; e < 16; ++e) bf[nj][e] = Bsp[nn * BSTR + kb + e];
    }
#pragma unroll
    for (int mi = 0; mi < 2; ++mi)
#pragma unroll
      for (int nj = 0; nj < 4; ++nj)
        acc[mi][nj] = wmma_bf16(af[mi], bf[nj], acc[mi][nj]);

    if (more) store_tiles(p ^ 1);
    p ^= 1;
  }

#pragma unroll
  for (int mi = 0; mi < 2; ++mi)
#pragma unroll
    for (int nj = 0; nj < 4; ++nj)
#pragma unroll
      for (int r = 0; r < 8; ++r) {
        int row = m0 + wm + mi * 16 + hi * 8 + r;
        int col = n0 + wn + nj * 16 + ln;
        out[(size_t)row * 512 + col] = acc[mi][nj][r] + bias[col];
      }
}

extern "C" void kernel_launch(void* const* d_in, const int* in_sizes, int n_in,
                              void* d_out, int out_size, void* d_ws, size_t ws_size,
                              hipStream_t stream) {
  (void)in_sizes; (void)n_in; (void)out_size; (void)ws_size;
  const float* x      = (const float*)d_in[0];
  const float* mask   = (const float*)d_in[1];
  const float* qkv_w  = (const float*)d_in[2];
  const float* qkv_b  = (const float*)d_in[3];
  const float* proj_w = (const float*)d_in[4];
  const float* proj_b = (const float*)d_in[5];
  const float* bt     = (const float*)d_in[6];
  float* out = (float*)d_out;

  const size_t msz = (size_t)2048 * TOK * 512;     // 51.4M (x / q / k / ao elems)
  const size_t vsz = (size_t)2048 * 16 * 32 * 64;  // 67.1M (vT elems, padded)
  char* base = (char*)d_ws;
  size_t off = 0;
  auto carve = [&](size_t bytes) { char* p = base + off; off += (bytes + 255) & ~(size_t)255; return p; };
  __bf16* xb     = (__bf16*)carve(msz * 2);
  __bf16* wqkvT  = (__bf16*)carve((size_t)512 * 1536 * 2);
  __bf16* wprojT = (__bf16*)carve((size_t)512 * 512 * 2);
  __bf16* qb     = (__bf16*)carve(msz * 2);
  __bf16* kb     = (__bf16*)carve(msz * 2);
  __bf16* vT     = (__bf16*)carve(vsz * 2);
  __bf16* ao     = (__bf16*)carve(msz * 2);
  float*  Cb     = (float*)carve((size_t)64 * 16 * TOK * TOK * 4);

  cvt4_f32_bf16<<<(int)(msz / 4 / 256), 256, 0, stream>>>(x, xb, (int)(msz / 4));
  cvt_wT<512, 1536><<<dim3(6, 512), 256, 0, stream>>>(qkv_w, wqkvT);
  cvt_wT<512, 512><<<dim3(2, 512), 256, 0, stream>>>(proj_w, wprojT);
  build_bias<<<dim3(10, 1024), 256, 0, stream>>>(mask, bt, Cb);
  zero_vt_pad<<<(2048 * 16 * 32 + 255) / 256, 256, 0, stream>>>(vT, 2048 * 16 * 32);
  qkv_gemm<<<dim3(12, 1568), 128, 0, stream>>>(xb, wqkvT, qkv_b, qb, kb, vT);
  attn_kernel<<<8192, 128, 0, stream>>>(qb, kb, vT, Cb, ao);
  proj_gemm<<<dim3(4, 1568), 128, 0, stream>>>(ao, wprojT, proj_b, out);
}
